// MultiHeadAttention_82300163326596
// MI455X (gfx1250) — matrix operand order repaired and checked by reading, not compiled, not run on hardware
//
#include <hip/hip_runtime.h>

// MI455X / gfx1250 wave32 WMMA implementation of fused MHA forward.
typedef __attribute__((ext_vector_type(16))) _Float16 v16h;
typedef __attribute__((ext_vector_type(8)))  _Float16 v8h;
typedef __attribute__((ext_vector_type(8)))  float    v8f;
typedef __attribute__((ext_vector_type(4)))  int      v4i;

#define BB 2
#define TT 2048
#define DD 1024
#define HH 16
#define HD 64
#define BT (BB*TT)   // 4096 rows for the projection GEMMs

// gfx1250 async global->LDS copies (ASYNCcnt), if the toolchain exposes them.
#if defined(__has_builtin)
#  if __has_builtin(__builtin_amdgcn_global_load_async_to_lds_b128) && \
      __has_builtin(__builtin_amdgcn_s_wait_asynccnt)
#    define HAVE_ASYNC_LDS 1
#  endif
#endif
#ifndef HAVE_ASYNC_LDS
#  define HAVE_ASYNC_LDS 0
#endif

#if HAVE_ASYNC_LDS
// Builtin signature (from hipcc diagnostic): (int4 AS1*, int4 AS3*, Ii, Ii).
typedef __attribute__((address_space(1))) v4i as1_v4i;
typedef __attribute__((address_space(3))) v4i as3_v4i;
#endif

// 16x32 f16 A-fragment in contiguous k order (B uses the same): per lane, elements 0..15 are K = hl*16+0..15
// ->  two contiguous 16-byte runs.
__device__ __forceinline__ v16h load_a_frag(const _Float16* p, int hl) {
  v8h lo = *(const v8h*)(p + hl * 16);
  v8h hi = *(const v8h*)(p + hl * 16 + 8);
  return __builtin_shufflevector(lo, hi, 0,1,2,3,4,5,6,7,8,9,10,11,12,13,14,15);
}

// 16 contiguous halves (two 16B runs) -> v16h (used for B fragments).
__device__ __forceinline__ v16h load_b_frag16(const _Float16* p) {
  v8h lo = *(const v8h*)(p);
  v8h hi = *(const v8h*)(p + 8);
  return __builtin_shufflevector(lo, hi, 0,1,2,3,4,5,6,7,8,9,10,11,12,13,14,15);
}

// ---------------------------------------------------------------------------
// Phase 0: bulk f32 -> f16 conversion (inputs and weights), 8 elems/thread.
// ---------------------------------------------------------------------------
__global__ __launch_bounds__(256)
void cvt_kernel(const float* __restrict__ src, _Float16* __restrict__ dst, int n) {
  const int i = (blockIdx.x * 256 + threadIdx.x) * 8;
  if (i >= n) return;
  const float4 a = *(const float4*)(src + i);
  const float4 b = *(const float4*)(src + i + 4);
  v8h o;
  o[0] = (_Float16)a.x; o[1] = (_Float16)a.y; o[2] = (_Float16)a.z; o[3] = (_Float16)a.w;
  o[4] = (_Float16)b.x; o[5] = (_Float16)b.y; o[6] = (_Float16)b.z; o[7] = (_Float16)b.w;
  *(v8h*)(dst + i) = o;
}

// ---------------------------------------------------------------------------
// WMMA GEMM: out = A @ W^T + bias.  A [4096,1024] f16, W [1024,1024] f16.
// 256 thr = 8 waves (4M x 2N), block tile 64x128, each wave 16x64.
// W tile (128x32 f16, 8KB) is staged to LDS with double-buffered async
// global->LDS b128 copies (ASYNCcnt); A-frags stream direct from global.
// layout 0: f16 [B,H,T,HD] (*scale)   layout 1: f16 [B,H,HD,T] (V transposed)
// layout 2: f32 [B,T,D] final output
// ---------------------------------------------------------------------------
__global__ __launch_bounds__(256)
void gemm_wmma_kernel(const _Float16* __restrict__ A, const _Float16* __restrict__ W,
                      const float* __restrict__ bias, const float scale,
                      const int layout, _Float16* __restrict__ outh,
                      float* __restrict__ outf)
{
  const int lane = threadIdx.x & 31;
  const int ln = lane & 15, hl = lane >> 4;
  const int wid = threadIdx.x >> 5;
  const int wm = wid & 3, wn = wid >> 2;

  const int row0 = blockIdx.x * 64 + wm * 16;    // token rows
  const int col0 = blockIdx.y * 128 + wn * 64;   // output features (this wave)
  const int colB = blockIdx.y * 128;             // output features (block)

  const v8f vzero = {0.f,0.f,0.f,0.f,0.f,0.f,0.f,0.f};
  v8f acc[4];
#pragma unroll
  for (int t = 0; t < 4; ++t) acc[t] = vzero;

  const _Float16* arow = A + (size_t)(row0 + ln) * DD;

#if HAVE_ASYNC_LDS
  __shared__ _Float16 ldsW[2][128][40];   // 40-half row stride: 16B aligned,
                                          // conflict-free b128 ds reads
  const int srow  = threadIdx.x >> 1;          // 0..127 tile row per thread
  const int shalf = (threadIdx.x & 1) * 16;    // which 16-half chunk
  const _Float16* wsrc = W + (size_t)(colB + srow) * DD + shalf;

  auto stage = [&](int buf, int kb) {
    _Float16* lp = &ldsW[buf][srow][shalf];
    const _Float16* gp = wsrc + kb;
    __builtin_amdgcn_global_load_async_to_lds_b128((as1_v4i*)gp,       (as3_v4i*)lp,       0, 0);
    __builtin_amdgcn_global_load_async_to_lds_b128((as1_v4i*)(gp + 8), (as3_v4i*)(lp + 8), 0, 0);
  };

  stage(0, 0);
#pragma unroll 2
  for (int kc = 0; kc < DD / 32; ++kc) {
    const int cur = kc & 1;
    __builtin_amdgcn_s_wait_asynccnt(0);
    __syncthreads();                              // tile[cur] resident in LDS
    if (kc + 1 < DD / 32) stage(1 - cur, (kc + 1) * 32);
    const v16h a  = load_a_frag(arow + kc * 32, hl);
    const _Float16* lrow = &ldsW[cur][wn * 64 + ln][hl * 16];
    const v16h b0 = load_b_frag16(lrow + 0  * 40);
    const v16h b1 = load_b_frag16(lrow + 16 * 40);
    const v16h b2 = load_b_frag16(lrow + 32 * 40);
    const v16h b3 = load_b_frag16(lrow + 48 * 40);
    acc[0] = __builtin_amdgcn_wmma_f32_16x16x32_f16(false, a, false, b0, (short)0, acc[0], false, false);
    acc[1] = __builtin_amdgcn_wmma_f32_16x16x32_f16(false, a, false, b1, (short)0, acc[1], false, false);
    acc[2] = __builtin_amdgcn_wmma_f32_16x16x32_f16(false, a, false, b2, (short)0, acc[2], false, false);
    acc[3] = __builtin_amdgcn_wmma_f32_16x16x32_f16(false, a, false, b3, (short)0, acc[3], false, false);
    __syncthreads();                              // reads done before overwrite
  }
#else
  const _Float16* wrow0 = W + (size_t)(col0 + ln) * DD + hl * 16;  // B(k,n)=W[n][k]
#pragma unroll 2
  for (int kc = 0; kc < DD / 32; ++kc) {
    const int kb = kc * 32;
    const v16h a  = load_a_frag(arow + kb, hl);
    const v16h b0 = *(const v16h*)(wrow0 + (size_t)0  * DD + kb);
    const v16h b1 = *(const v16h*)(wrow0 + (size_t)16 * DD + kb);
    const v16h b2 = *(const v16h*)(wrow0 + (size_t)32 * DD + kb);
    const v16h b3 = *(const v16h*)(wrow0 + (size_t)48 * DD + kb);
    acc[0] = __builtin_amdgcn_wmma_f32_16x16x32_f16(false, a, false, b0, (short)0, acc[0], false, false);
    acc[1] = __builtin_amdgcn_wmma_f32_16x16x32_f16(false, a, false, b1, (short)0, acc[1], false, false);
    acc[2] = __builtin_amdgcn_wmma_f32_16x16x32_f16(false, a, false, b2, (short)0, acc[2], false, false);
    acc[3] = __builtin_amdgcn_wmma_f32_16x16x32_f16(false, a, false, b3, (short)0, acc[3], false, false);
  }
#endif

#pragma unroll
  for (int nt = 0; nt < 4; ++nt) {
    const int o = col0 + nt * 16 + ln;
    const int hh = o >> 6, hd = o & 63;
    const float bia = bias[o];
#pragma unroll
    for (int i = 0; i < 8; ++i) {
      const int r = row0 + hl * 8 + i;          // global row in [0, B*T)
      const int bb = r >> 11, t = r & (TT - 1);
      const float val = (acc[nt][i] + bia) * scale;
      if (layout == 0)
        outh[(((size_t)(bb * HH + hh)) * TT + t) * HD + hd] = (_Float16)val;
      else if (layout == 1)
        outh[(((size_t)(bb * HH + hh)) * HD + hd) * TT + t] = (_Float16)val;
      else
        outf[(size_t)r * DD + o] = val;
    }
  }
}

// ---------------------------------------------------------------------------
// Phase 2: causal flash attention with bias. 128 threads = 4 waves; each wave
// owns one 16-row query tile and loops over 32-key blocks (causal-trimmed).
// ---------------------------------------------------------------------------
__global__ __launch_bounds__(128)
void flash_attn_kernel(const _Float16* __restrict__ qf, const _Float16* __restrict__ kf,
                       const _Float16* __restrict__ vT,
                       const float* __restrict__ attn_bias,
                       const unsigned char* __restrict__ kpm,
                       _Float16* __restrict__ attn)
{
  __shared__ _Float16 ldsP[4][16][40];   // per-wave 16x32 P tile, 80B row stride

  const int lane = threadIdx.x & 31;
  const int ln = lane & 15, hl = lane >> 4;
  const int wid = threadIdx.x >> 5;

  const int bh = blockIdx.x >> 5;        // 32 q-groups per (b,h)
  const int qg = blockIdx.x & 31;
  const int b = bh >> 4, h = bh & 15;
  const int q0 = (qg * 4 + wid) * 16;

  const size_t headQK = (size_t)(b * HH + h) * TT;

  // Q A-fragments: 16x64 as two 16x32 K-chunks (vectorized b128 loads)
  const _Float16* qrow = qf + (headQK + q0 + ln) * HD;
  v16h aq[2];
#pragma unroll
  for (int ck = 0; ck < 2; ++ck) aq[ck] = load_a_frag(qrow + ck * 32, hl);

  const v8f vzero = {0.f,0.f,0.f,0.f,0.f,0.f,0.f,0.f};
  float m[8], l[8];
  v8f O[4];
#pragma unroll
  for (int i = 0; i < 8; ++i) { m[i] = -1e30f; l[i] = 0.f; }
#pragma unroll
  for (int t = 0; t < 4; ++t) O[t] = vzero;

  const float* biasBase = attn_bias + ((size_t)(b * HH + h)) * TT * TT;
  const int kb_end = (q0 + 16 + 31) >> 5;          // causal: keys <= q0+15

  for (int kb = 0; kb < kb_end; ++kb) {
    const int key0 = kb * 32;
    float S[2][8];
#pragma unroll
    for (int s = 0; s < 2; ++s) {
      v8f c = vzero;
      const _Float16* krow = kf + (headQK + key0 + s * 16 + ln) * HD + hl * 16;
      const v16h bk0 = *(const v16h*)(krow);        // B(k=d, n=key)
      const v16h bk1 = *(const v16h*)(krow + 32);
      c = __builtin_amdgcn_wmma_f32_16x16x32_f16(false, aq[0], false, bk0, (short)0, c, false, false);
      c = __builtin_amdgcn_wmma_f32_16x16x32_f16(false, aq[1], false, bk1, (short)0, c, false, false);
      const int key = key0 + s * 16 + ln;
      const bool kmask = kpm[b * TT + key] != 0;
#pragma unroll
      for (int i = 0; i < 8; ++i) {
        const int r = q0 + hl * 8 + i;
        // 512 MiB single-use stream: keep it out of the caches.
        float sc = c[i] + __builtin_nontemporal_load(biasBase + (size_t)r * TT + key);
        if (key > r || kmask) sc = -1e30f;
        S[s][i] = sc;
      }
    }

    // Prefetch next key-block's bias rows (lowers to global_prefetch_b8).
    if (kb + 1 < kb_end) {
#pragma unroll
      for (int i = 0; i < 8; ++i) {
        const int r = q0 + hl * 8 + i;
        __builtin_prefetch(biasBase + (size_t)r * TT + key0 + 32 + (ln & 7) * 4, 0, 0);
      }
    }

    // Online softmax: row reductions across the 16 lanes of each wave half.
    float p0[8], p1[8];
#pragma unroll
    for (int i = 0; i < 8; ++i) {
      float mx = fmaxf(S[0][i], S[1][i]);
      for (int off = 1; off < 16; off <<= 1) mx = fmaxf(mx, __shfl_xor(mx, off, 32));
      const float nm = fmaxf(m[i], mx);
      const float corr = __expf(m[i] - nm);
      p0[i] = __expf(S[0][i] - nm);
      p1[i] = __expf(S[1][i] - nm);
      float rs = p0[i] + p1[i];
      for (int off = 1; off < 16; off <<= 1) rs += __shfl_xor(rs, off, 32);
      l[i] = l[i] * corr + rs;
      m[i] = nm;
#pragma unroll
      for (int t = 0; t < 4; ++t) O[t][i] *= corr;
    }

    // P: C-layout regs -> LDS -> A-layout fragment (in-order DS within wave).
#pragma unroll
    for (int i = 0; i < 8; ++i) {
      ldsP[wid][hl * 8 + i][ln]      = (_Float16)p0[i];
      ldsP[wid][hl * 8 + i][16 + ln] = (_Float16)p1[i];
    }
    __builtin_amdgcn_wave_barrier();
    const _Float16* prow = &ldsP[wid][ln][0];
    const v8h plo = *(const v8h*)(prow + hl * 16);
    const v8h phi = *(const v8h*)(prow + hl * 16 + 8);
    const v16h pa = __builtin_shufflevector(plo, phi, 0,1,2,3,4,5,6,7,8,9,10,11,12,13,14,15);
    __builtin_amdgcn_wave_barrier();

    // O += P x V : V stored transposed, so each B-frag is 32B contiguous.
    const _Float16* vrow = vT + ((size_t)(b * HH + h) * HD) * TT + key0 + hl * 16;
    const v16h bv0 = *(const v16h*)(vrow + (size_t)(0  + ln) * TT);
    const v16h bv1 = *(const v16h*)(vrow + (size_t)(16 + ln) * TT);
    const v16h bv2 = *(const v16h*)(vrow + (size_t)(32 + ln) * TT);
    const v16h bv3 = *(const v16h*)(vrow + (size_t)(48 + ln) * TT);
    O[0] = __builtin_amdgcn_wmma_f32_16x16x32_f16(false, pa, false, bv0, (short)0, O[0], false, false);
    O[1] = __builtin_amdgcn_wmma_f32_16x16x32_f16(false, pa, false, bv1, (short)0, O[1], false, false);
    O[2] = __builtin_amdgcn_wmma_f32_16x16x32_f16(false, pa, false, bv2, (short)0, O[2], false, false);
    O[3] = __builtin_amdgcn_wmma_f32_16x16x32_f16(false, pa, false, bv3, (short)0, O[3], false, false);
  }

  // Normalize and emit f16 attention output [B,T,D] (head-merged).
#pragma unroll
  for (int i = 0; i < 8; ++i) {
    const float inv = 1.0f / l[i];
    const int r = q0 + hl * 8 + i;
    _Float16* dst = attn + ((size_t)b * TT + r) * DD + h * HD;
#pragma unroll
    for (int t = 0; t < 4; ++t) dst[t * 16 + ln] = (_Float16)(O[t][i] * inv);
  }
}

// ---------------------------------------------------------------------------
extern "C" void kernel_launch(void* const* d_in, const int* in_sizes, int n_in,
                              void* d_out, int out_size, void* d_ws, size_t ws_size,
                              hipStream_t stream) {
  (void)in_sizes; (void)n_in; (void)out_size; (void)ws_size;
  const float* query      = (const float*)d_in[0];
  const float* key        = (const float*)d_in[1];
  const float* value      = (const float*)d_in[2];
  const float* attn_bias  = (const float*)d_in[3];
  const unsigned char* kpm = (const unsigned char*)d_in[4];
  const float* wq = (const float*)d_in[5];
  const float* bq = (const float*)d_in[6];
  const float* wk = (const float*)d_in[7];
  const float* bk = (const float*)d_in[8];
  const float* wv = (const float*)d_in[9];
  const float* bv = (const float*)d_in[10];
  const float* wo = (const float*)d_in[11];
  const float* bo = (const float*)d_in[12];
  float* out = (float*)d_out;

  const size_t NX = (size_t)BT * DD;        // 4 Mi elements per activation
  const size_t NW = (size_t)DD * DD;        // 1 Mi elements per weight
  _Float16* xq  = (_Float16*)d_ws;          // f16 copies of inputs
  _Float16* xk  = xq  + NX;
  _Float16* xv  = xk  + NX;
  _Float16* wqh = xv  + NX;                 // f16 weights
  _Float16* wkh = wqh + NW;
  _Float16* wvh = wkh + NW;
  _Float16* woh = wvh + NW;
  _Float16* qf  = woh + NW;                 // [B,H,T,HD]
  _Float16* kf  = qf  + NX;                 // [B,H,T,HD]
  _Float16* vT  = kf  + NX;                 // [B,H,HD,T]
  _Float16* attn = xq;                      // reuse: xq retired after Q proj

  // Phase 0: convert inputs + weights to f16.
  const int cb = 256 * 8;
  cvt_kernel<<<dim3((int)(NX / cb)), 256, 0, stream>>>(query, xq, (int)NX);
  cvt_kernel<<<dim3((int)(NX / cb)), 256, 0, stream>>>(key,   xk, (int)NX);
  cvt_kernel<<<dim3((int)(NX / cb)), 256, 0, stream>>>(value, xv, (int)NX);
  cvt_kernel<<<dim3((int)(NW / cb)), 256, 0, stream>>>(wq, wqh, (int)NW);
  cvt_kernel<<<dim3((int)(NW / cb)), 256, 0, stream>>>(wk, wkh, (int)NW);
  cvt_kernel<<<dim3((int)(NW / cb)), 256, 0, stream>>>(wv, wvh, (int)NW);
  cvt_kernel<<<dim3((int)(NW / cb)), 256, 0, stream>>>(wo, woh, (int)NW);

  // Phase 1: Q/K/V projections (Q pre-scaled by HD^-0.5, V transposed).
  dim3 gg(BT / 64, DD / 128);
  gemm_wmma_kernel<<<gg, 256, 0, stream>>>(xq, wqh, bq, 0.125f, 0, qf, nullptr);
  gemm_wmma_kernel<<<gg, 256, 0, stream>>>(xk, wkh, bk, 1.0f,   0, kf, nullptr);
  gemm_wmma_kernel<<<gg, 256, 0, stream>>>(xv, wvh, bv, 1.0f,   1, vT, nullptr);

  // Phase 2: flash attention.
  flash_attn_kernel<<<dim3(BB * HH * (TT / 64)), 128, 0, stream>>>(
      qf, kf, vT, attn_bias, kpm, attn);

  // Phase 3: output projection -> f32.
  gemm_wmma_kernel<<<gg, 256, 0, stream>>>(attn, woh, bo, 1.0f, 2, nullptr, out);
}
